// EfficientAttention_76819785056901
// MI455X (gfx1250) — matrix operand order repaired and checked by reading, not compiled, not run on hardware
//
#include <hip/hip_runtime.h>

typedef __attribute__((ext_vector_type(16))) _Float16 v16h;
typedef __attribute__((ext_vector_type(8)))  float    v8f;

#define WMMA_F16(A, B, C) \
  __builtin_amdgcn_wmma_f32_16x16x32_f16(false, (A), false, (B), (short)0, (C), false, false)

constexpr int Bsz  = 4;
constexpr int SEQ  = 4096;
constexpr int Hn   = 8;
constexpr int Dh   = 64;
constexpr int ROWS = Bsz * SEQ * Hn;   // 131072

// ---- async memory->LDS staging (gfx1250 ASYNCcnt path with sync fallback) ---
#if __has_builtin(__builtin_amdgcn_global_load_async_to_lds_b128)
#define HAVE_ASYNC_LDS 1
// param type per compiler diagnostic: vector-of-4-int pointers in AS1 / AS3
typedef __attribute__((__vector_size__(4 * sizeof(int)))) int v4i;
typedef __attribute__((address_space(1))) v4i* g_v4i_ptr;
typedef __attribute__((address_space(3))) v4i* s_v4i_ptr;
#endif

__device__ inline void wait_async0() {
#if __has_builtin(__builtin_amdgcn_s_wait_asynccnt)
  __builtin_amdgcn_s_wait_asynccnt(0);
#else
  asm volatile("s_wait_asynccnt 0x0" ::: "memory");
#endif
}

// copy one contiguous 8192-byte tile (64x64 f16) global -> LDS with 256 threads
__device__ inline void stage8k(const _Float16* __restrict__ g, _Float16* __restrict__ s) {
  const int t = threadIdx.x;
#if HAVE_ASYNC_LDS
  g_v4i_ptr gp = (g_v4i_ptr)(g + (size_t)t * 8);          // 16B per thread
  s_v4i_ptr sp = (s_v4i_ptr)(s + (size_t)t * 8);
  __builtin_amdgcn_global_load_async_to_lds_b128(gp, sp, 0, 0);
  __builtin_amdgcn_global_load_async_to_lds_b128(gp + 256, sp + 256, 0, 0);
#else
  const uint4* gv = (const uint4*)g;
  uint4* sv = (uint4*)s;
  sv[t] = gv[t];
  sv[t + 256] = gv[t + 256];
#endif
}

// stage a 64x64 f16 tile whose 64 global rows (d) have stride SEQ halfs
// (V^T layout [d][n]); LDS result is dense [d][64]
__device__ inline void stage_vtile(const _Float16* __restrict__ g, _Float16* __restrict__ s) {
  const int t = threadIdx.x;
#pragma unroll
  for (int i = 0; i < 2; ++i) {
    const int c  = t + i * 256;     // 512 chunks of 16B
    const int d  = c >> 3;
    const int ko = (c & 7) * 8;
#if HAVE_ASYNC_LDS
    g_v4i_ptr gp = (g_v4i_ptr)(g + (size_t)d * SEQ + ko);
    s_v4i_ptr sp = (s_v4i_ptr)(s + d * 64 + ko);
    __builtin_amdgcn_global_load_async_to_lds_b128(gp, sp, 0, 0);
#else
    *(uint4*)(s + d * 64 + ko) = *(const uint4*)(g + (size_t)d * SEQ + ko);
#endif
  }
}

// ---- fragment loaders -------------------------------------------------------
// A 16x32 f16 layout: lanes 0-15 hold row M=lane, halfs j=0..15 -> K=kbase+lhalf*16+j
// (lanes 16-31: K=kbase+16..31), matching the B loader's contiguous K order.
__device__ inline v16h load_a_f32(const float* __restrict__ row, int kbase, int lhalf) {
  v16h a;
#pragma unroll
  for (int j = 0; j < 8; ++j) {
    a[j]     = (_Float16)row[kbase + lhalf * 16 + j];
    a[j + 8] = (_Float16)row[kbase + lhalf * 16 + 8 + j];
  }
  return a;
}
__device__ inline v16h load_a_f16(const _Float16* __restrict__ row, int kbase, int lhalf) {
  v16h a;
#pragma unroll
  for (int j = 0; j < 8; ++j) {
    a[j]     = row[kbase + lhalf * 16 + j];
    a[j + 8] = row[kbase + lhalf * 16 + 8 + j];
  }
  return a;
}
// B 32x16 f16: lane n holds column N=n&15; halfs j=0..15 -> K = kbase+lhalf*16+j.
// With an n-major (transposed) f16 source these 16 K's are contiguous.
__device__ inline v16h load_b_lds_row(const _Float16* __restrict__ p) {
  v16h b;
#pragma unroll
  for (int j = 0; j < 16; ++j) b[j] = p[j];
  return b;
}

// ---- kernel 1: fused q/k/v projections, f32 -> f16 --------------------------
// q,k stored [b][h][n][d]; v stored transposed [b][h][d][n] for the PV gemm
__global__ __launch_bounds__(128) void ea_proj_kernel(
    const float* __restrict__ x,
    const float* __restrict__ Wq, const float* __restrict__ Wk, const float* __restrict__ Wv,
    _Float16* __restrict__ qws, _Float16* __restrict__ kws, _Float16* __restrict__ vws) {
  __shared__ __align__(16) _Float16 WtT[64 * 64];   // W^T as f16: WtT[f*64 + k]

  const int lane  = threadIdx.x & 31;
  const int wv    = threadIdx.x >> 5;
  const int l16   = lane & 15;
  const int lhalf = lane >> 4;
  const int tile  = blockIdx.x * 4 + wv;            // 16-row tile over B*N*H rows
  const int m     = blockIdx.y;                     // 0=q 1=k 2=v

  const float* W   = (m == 0) ? Wq : (m == 1) ? Wk : Wv;
  _Float16*    dst = (m == 0) ? qws : (m == 1) ? kws : vws;
  const float  scl = (m == 0) ? 0.125f : 1.0f;      // fold 1/sqrt(64) into q
  const bool   vT  = (m == 2);

  {  // stage W^T once per block: thread t -> row k=t>>1, 32 cols
    const int t = threadIdx.x;
    const int k = t >> 1, f0 = (t & 1) * 32;
    const float* wr = W + k * 64 + f0;
#pragma unroll
    for (int j = 0; j < 32; ++j) WtT[(f0 + j) * 64 + k] = (_Float16)wr[j];
  }
  __syncthreads();

  const int    r0   = tile * 16;
  const float* rowp = x + (size_t)(r0 + l16) * Dh;
  v16h a0 = load_a_f32(rowp, 0, lhalf);
  v16h a1 = load_a_f32(rowp, 32, lhalf);

#pragma unroll
  for (int nt = 0; nt < 4; ++nt) {
    const int col = nt * 16 + l16;
    const _Float16* wcol = WtT + col * 64 + lhalf * 16;
    v16h b0 = load_b_lds_row(wcol);
    v16h b1 = load_b_lds_row(wcol + 32);
    v8f acc = {};
    acc = WMMA_F16(a0, b0, acc);
    acc = WMMA_F16(a1, b1, acc);
#pragma unroll
    for (int v = 0; v < 8; ++v) {
      const int r2 = r0 + v + 8 * lhalf;            // row in [b][n][h] flattening
      const int h  = r2 & 7;
      const int bn = r2 >> 3;
      const int n  = bn & (SEQ - 1);
      const int b  = bn >> 12;
      const size_t bh = (size_t)(b * Hn + h);
      const size_t idx = vT ? (bh * Dh + col) * SEQ + n
                            : (bh * SEQ + n) * Dh + col;
      dst[idx] = (_Float16)(acc[v] * scl);
    }
  }
}

// ---- kernel 2: flash attention + fused output projection --------------------
__global__ __launch_bounds__(256) void ea_attn_kernel(
    const _Float16* __restrict__ qws, const _Float16* __restrict__ kws,
    const _Float16* __restrict__ vws,
    const float* __restrict__ Wo, const float* __restrict__ bo,
    float* __restrict__ out) {
  __shared__ __align__(16) _Float16 Klds[2][64 * 64];   // double-buffered K tile [key][d]
  __shared__ __align__(16) _Float16 Vlds[2][64 * 64];   // double-buffered V^T tile [d][key]
  __shared__ __align__(16) _Float16 Plds[8 * 16 * 64];  // per-wave C->A staging

  const int lane  = threadIdx.x & 31;
  const int wv    = threadIdx.x >> 5;
  const int l16   = lane & 15;
  const int lhalf = lane >> 4;
  const int bh    = blockIdx.y;
  const int bb    = bh >> 3;
  const int hh    = bh & 7;
  const int q0    = blockIdx.x * 128 + wv * 16;     // this wave's 16 query rows

  const _Float16* Q  = qws + (size_t)bh * SEQ * Dh;
  const _Float16* K  = kws + (size_t)bh * SEQ * Dh;
  const _Float16* Vt = vws + (size_t)bh * Dh * SEQ; // [d][n]

  const _Float16* qrow = Q + (size_t)(q0 + l16) * Dh;
  v16h aQ0 = load_a_f16(qrow, 0, lhalf);
  v16h aQ1 = load_a_f16(qrow, 32, lhalf);

  v8f O[4];
  float mrow[8], lrow[8];
#pragma unroll
  for (int nt = 0; nt < 4; ++nt)
#pragma unroll
    for (int v = 0; v < 8; ++v) O[nt][v] = 0.0f;
#pragma unroll
  for (int v = 0; v < 8; ++v) { mrow[v] = -3.0e38f; lrow[v] = 0.0f; }

  _Float16* Pw = Plds + wv * 16 * 64;

  // prologue: stage first K/V tile
  stage8k(K, Klds[0]);
  stage_vtile(Vt, Vlds[0]);
  wait_async0();
  __syncthreads();

  constexpr int NT = SEQ / 64;
  for (int kt = 0; kt < NT; ++kt) {
    const int cur = kt & 1;
    if (kt + 1 < NT) {          // async prefetch of next tile into the other buffer
      stage8k(K + (size_t)(kt + 1) * 64 * Dh, Klds[cur ^ 1]);
      stage_vtile(Vt + (size_t)(kt + 1) * 64, Vlds[cur ^ 1]);
    }

    // S(16x64) = Q(16x64) @ K^T ; B_{k=d, n=key} is a contiguous LDS row slice
    v8f S[4];
#pragma unroll
    for (int nt = 0; nt < 4; ++nt) {
      const _Float16* kr = Klds[cur] + (nt * 16 + l16) * Dh + lhalf * 16;
      v16h b0 = load_b_lds_row(kr);
      v16h b1 = load_b_lds_row(kr + 32);
      v8f s = {};
      s = WMMA_F16(aQ0, b0, s);
      s = WMMA_F16(aQ1, b1, s);
      S[nt] = s;
    }

    // online softmax; row r = v + 8*lhalf lives in the lanes of this half
#pragma unroll
    for (int v = 0; v < 8; ++v) {
      float mx = fmaxf(fmaxf(S[0][v], S[1][v]), fmaxf(S[2][v], S[3][v]));
#pragma unroll
      for (int off = 1; off < 16; off <<= 1) mx = fmaxf(mx, __shfl_xor(mx, off, 32));
      const float mn   = fmaxf(mrow[v], mx);
      const float corr = __expf(mrow[v] - mn);
      const int   row  = v + 8 * lhalf;
      float psum = 0.0f;
#pragma unroll
      for (int nt = 0; nt < 4; ++nt) {
        const float p = __expf(S[nt][v] - mn);
        psum += p;
        Pw[row * 64 + nt * 16 + l16] = (_Float16)p;
        O[nt][v] *= corr;
      }
#pragma unroll
      for (int off = 1; off < 16; off <<= 1) psum += __shfl_xor(psum, off, 32);
      lrow[v] = lrow[v] * corr + psum;
      mrow[v] = mn;
    }
    __syncthreads();  // make this wave's P writes visible across its lanes

    // O(16x64) += P(16x64) @ V(64x64); V^T tile gives contiguous B slices
    v16h pa0 = load_a_f16(Pw + l16 * 64, 0, lhalf);
    v16h pa1 = load_a_f16(Pw + l16 * 64, 32, lhalf);
#pragma unroll
    for (int nt = 0; nt < 4; ++nt) {
      const _Float16* vr = Vlds[cur] + (nt * 16 + l16) * 64 + lhalf * 16;
      v16h vb0 = load_b_lds_row(vr);
      v16h vb1 = load_b_lds_row(vr + 32);
      O[nt] = WMMA_F16(pa0, vb0, O[nt]);
      O[nt] = WMMA_F16(pa1, vb1, O[nt]);
    }

    wait_async0();    // next tile resident
    __syncthreads();  // all waves done reading current tiles
  }

  // normalize, restage hidden (f16) into Pw, and stage Wo^T into dead K buffer
#pragma unroll
  for (int v = 0; v < 8; ++v) {
    const float inv = 1.0f / lrow[v];
    const int   row = v + 8 * lhalf;
#pragma unroll
    for (int nt = 0; nt < 4; ++nt)
      Pw[row * 64 + nt * 16 + l16] = (_Float16)(O[nt][v] * inv);
  }
  _Float16* WoT = Klds[0];    // 8KB, no longer needed
  {
    const int t = threadIdx.x;            // 256 threads
    const int k = t >> 2, f0 = (t & 3) * 16;
    const float* wr = Wo + k * 64 + f0;
#pragma unroll
    for (int j = 0; j < 16; ++j) WoT[(f0 + j) * 64 + k] = (_Float16)wr[j];
  }
  __syncthreads();

  // epilogue: out = hidden @ Wo + bo, written to [b][n][h][d]
  v16h ha0 = load_a_f16(Pw + l16 * 64, 0, lhalf);
  v16h ha1 = load_a_f16(Pw + l16 * 64, 32, lhalf);
#pragma unroll
  for (int nt = 0; nt < 4; ++nt) {
    const int col = nt * 16 + l16;
    const _Float16* wcol = WoT + col * 64 + lhalf * 16;
    v16h wb0 = load_b_lds_row(wcol);
    v16h wb1 = load_b_lds_row(wcol + 32);
    const float bv = bo[col];
    v8f acc = { bv, bv, bv, bv, bv, bv, bv, bv };
    acc = WMMA_F16(ha0, wb0, acc);
    acc = WMMA_F16(ha1, wb1, acc);
#pragma unroll
    for (int v = 0; v < 8; ++v) {
      const int n = q0 + v + 8 * lhalf;
      out[(((size_t)bb * SEQ + n) * Hn + hh) * Dh + col] = acc[v];
    }
  }
}

// ---- launcher ---------------------------------------------------------------
extern "C" void kernel_launch(void* const* d_in, const int* in_sizes, int n_in,
                              void* d_out, int out_size, void* d_ws, size_t ws_size,
                              hipStream_t stream) {
  const float* x  = (const float*)d_in[0];
  const float* Wq = (const float*)d_in[1];
  const float* Wk = (const float*)d_in[2];
  const float* Wv = (const float*)d_in[3];
  const float* Wo = (const float*)d_in[4];
  const float* bo = (const float*)d_in[5];
  float* out = (float*)d_out;

  _Float16* qws = (_Float16*)d_ws;                    // [b][h][n][d] f16, q pre-scaled
  _Float16* kws = qws + (size_t)ROWS * Dh;            // [b][h][n][d] f16
  _Float16* vws = kws + (size_t)ROWS * Dh;            // [b][h][d][n] f16 (transposed)

  dim3 pgrid(ROWS / 64, 3);                           // 4 waves/block, 16 rows/wave
  ea_proj_kernel<<<pgrid, 128, 0, stream>>>(x, Wq, Wk, Wv, qws, kws, vws);

  dim3 agrid(SEQ / 128, Bsz * Hn);                    // 128 q-rows/block, 32 (b,h)
  ea_attn_kernel<<<agrid, 256, 0, stream>>>(qws, kws, vws, Wo, bo, out);
}
